// LocalTransferNet_37898791420126
// MI455X (gfx1250) — matrix operand order repaired and checked by reading, not compiled, not run on hardware
//
#include <hip/hip_runtime.h>

typedef __attribute__((ext_vector_type(16))) _Float16 v16h;
typedef __attribute__((ext_vector_type(8)))  float    v8f;

#define TPB 256
#define KS  17

static inline int grid_for(long total) {
  long g = (total + TPB - 1) / TPB;
  if (g > 32768) g = 32768;
  if (g < 1) g = 1;
  return (int)g;
}

// ---------------------------------------------------------------------------
// concat(G_prev, G_cur) along channel dim -> [B,6,H,W]
// ---------------------------------------------------------------------------
__global__ void concat_kernel(const float* __restrict__ a, const float* __restrict__ b,
                              float* __restrict__ o, int Bn, int HW) {
  int total = Bn * 6 * HW;
  for (int i = blockIdx.x * blockDim.x + threadIdx.x; i < total; i += gridDim.x * blockDim.x) {
    int bb = i / (6 * HW);
    int r  = i - bb * 6 * HW;
    int c  = r / HW;
    int hw = r - c * HW;
    o[i] = (c < 3) ? a[((size_t)bb * 3 + c) * HW + hw]
                   : b[((size_t)bb * 3 + (c - 3)) * HW + hw];
  }
}

// ---------------------------------------------------------------------------
// 2x2 average pool : x[NC,H,W] -> y[NC,H/2,W/2]
// ---------------------------------------------------------------------------
__global__ void avgpool2_kernel(const float* __restrict__ x, float* __restrict__ y,
                                int NC, int H, int W) {
  int Ho = H >> 1, Wo = W >> 1;
  long total = (long)NC * Ho * Wo;
  for (long i = blockIdx.x * (long)blockDim.x + threadIdx.x; i < total; i += gridDim.x * (long)blockDim.x) {
    int n  = (int)(i / (Ho * Wo));
    int r  = (int)(i - (long)n * Ho * Wo);
    int ho = r / Wo;
    int wo = r - ho * Wo;
    const float* p = x + ((size_t)n * H + 2 * ho) * W + 2 * wo;
    y[i] = 0.25f * (p[0] + p[1] + p[W] + p[W + 1]);
  }
}

// ---------------------------------------------------------------------------
// nearest 2x upsample (+ optional skip add): x[NC,Ho/2,Wo/2] -> y[NC,Ho,Wo]
// ---------------------------------------------------------------------------
__global__ void upsample2_kernel(const float* __restrict__ x, const float* __restrict__ skip,
                                 float* __restrict__ y, int NC, int Ho, int Wo, int has_skip) {
  int Hi = Ho >> 1, Wi = Wo >> 1;
  long total = (long)NC * Ho * Wo;
  for (long i = blockIdx.x * (long)blockDim.x + threadIdx.x; i < total; i += gridDim.x * (long)blockDim.x) {
    int n = (int)(i / (Ho * Wo));
    int r = (int)(i - (long)n * Ho * Wo);
    int h = r / Wo;
    int w = r - h * Wo;
    float v = x[((size_t)n * Hi + (h >> 1)) * Wi + (w >> 1)];
    if (has_skip) v += skip[i];
    y[i] = v;
  }
}

// ---------------------------------------------------------------------------
// channel softmax (C small, in-place) over x[B,C,HW]
// ---------------------------------------------------------------------------
__global__ void softmax_c_kernel(float* __restrict__ x, int Bn, int C, int HW) {
  long total = (long)Bn * HW;
  for (long i = blockIdx.x * (long)blockDim.x + threadIdx.x; i < total; i += gridDim.x * (long)blockDim.x) {
    int bb = (int)(i / HW);
    int hw = (int)(i - (long)bb * HW);
    float* p = x + (size_t)bb * C * HW + hw;
    float mx = -1e30f;
    for (int c = 0; c < C; ++c) mx = fmaxf(mx, p[(size_t)c * HW]);
    float s = 0.f;
    for (int c = 0; c < C; ++c) { float e = __expf(p[(size_t)c * HW] - mx); p[(size_t)c * HW] = e; s += e; }
    float inv = 1.f / s;
    for (int c = 0; c < C; ++c) p[(size_t)c * HW] *= inv;
  }
}

// ---------------------------------------------------------------------------
// 17x17 separable adaptive filter with edge-replicate padding:
// out[b,c,h,w] = sum_u kh[b,u,h,w] * sum_v I[b,c,clamp(h+u-8),clamp(w+v-8)] * kw[b,v,h,w]
// ---------------------------------------------------------------------------
__global__ void sep_conv_kernel(const float* __restrict__ I, const float* __restrict__ kh,
                                const float* __restrict__ kw, float* __restrict__ out,
                                int Bn, int C, int H, int W) {
  int HW = H * W;
  long total = (long)Bn * C * HW;
  for (long i = blockIdx.x * (long)blockDim.x + threadIdx.x; i < total; i += gridDim.x * (long)blockDim.x) {
    int bb = (int)(i / (C * HW));
    int r  = (int)(i - (long)bb * C * HW);
    int c  = r / HW;
    int hw = r - c * HW;
    int h  = hw / W;
    int w  = hw - h * W;
    const float* khp = kh + (size_t)bb * KS * HW + hw;
    const float* kwp = kw + (size_t)bb * KS * HW + hw;
    float kwv[KS];
    #pragma unroll
    for (int v = 0; v < KS; ++v) kwv[v] = kwp[(size_t)v * HW];
    const float* img = I + ((size_t)bb * C + c) * HW;
    float acc = 0.f;
    for (int u = 0; u < KS; ++u) {
      int hh = h + u - 8; hh = hh < 0 ? 0 : (hh >= H ? H - 1 : hh);
      const float* row = img + (size_t)hh * W;
      float s = 0.f;
      #pragma unroll
      for (int v = 0; v < KS; ++v) {
        int ww = w + v - 8; ww = ww < 0 ? 0 : (ww >= W ? W - 1 : ww);
        s = fmaf(row[ww], kwv[v], s);
      }
      acc = fmaf(s, khp[(size_t)u * HW], acc);
    }
    out[i] = acc;
  }
}

// ---------------------------------------------------------------------------
// Pack fp32 weights w[Cout][Cin][3][3] into f16 WMMA A fragments with a
// tap-major, channel-padded K order:
//   chunk c (32 K values) -> tap = c / cpt, ic = (c % cpt)*32 + k32
//   (cpt = ceil(Cin/32); ic >= Cin padded with zeros)
// Within a chunk, A-fragment layout used here (contiguous k order, matched by B; 16-bit A 16x32):
//   half index = ((mt*nchunk + c)*32 + lane)*16 + j, row M = lane&15,
//   lane<16 : k32 = j                     // {0..15}
//   lane>=16: k32 = j+16                  // {16..31}
// ---------------------------------------------------------------------------
__global__ void pack_w_kernel(const float* __restrict__ W, _Float16* __restrict__ Ap,
                              int Cout, int Cin, int cpt, int mt_cnt, int nchunk) {
  long total = (long)mt_cnt * nchunk * 512;
  for (long idx = blockIdx.x * (long)blockDim.x + threadIdx.x; idx < total; idx += gridDim.x * (long)blockDim.x) {
    int j    = (int)(idx & 15);
    int lane = (int)((idx >> 4) & 31);
    long rest = idx >> 9;
    int c  = (int)(rest % nchunk);
    int mt = (int)(rest / nchunk);
    int m = lane & 15;
    int k32 = (lane < 16) ? j : j + 16;
    int tap = c / cpt;
    int ic  = (c - tap * cpt) * 32 + k32;
    int oc  = mt * 16 + m;
    float v = 0.f;
    if (oc < Cout && ic < Cin) v = W[((size_t)oc * Cin + ic) * 9 + tap];
    Ap[idx] = (_Float16)v;
  }
}

// ---------------------------------------------------------------------------
// Implicit-GEMM conv3x3 (SAME zero pad, stride 1) + bias + ReLU, via WMMA.
// K order is tap-major / channel-minor (matches pack_w_kernel), so each
// 32-wide K chunk has a single, chunk-invariant spatial tap:
//   - tap offset + zero-pad validity computed once per tap (unrolled x9)
//   - B elements are 16 consecutive input channels at one pixel: a clause of
//     independent strided loads, masked branchlessly by a 0/1 multiplier.
// CLAMP=false (Cin multiple of 32): element offsets are unsigned j*HW,
// loop-invariant -> hoisted to SGPRs, no per-element min/sign-extend chains.
// One wave computes MB (compile-time) m-subtiles of 16 oc x 16 pixels sharing
// each B fragment. fp32 in/out, f16 WMMA operands, f32 accumulation.
// ---------------------------------------------------------------------------
template <int MB, bool CLAMP>
__global__ __launch_bounds__(256) void conv3x3_wmma_kernel(
    const float* __restrict__ X, const _Float16* __restrict__ Ap,
    const float* __restrict__ bias, float* __restrict__ Y,
    int Bn, int Cin, int Cout, int H, int Wd,
    int cpt, int nchunk, int ntiles_n, int msuper) {
  int wid  = (blockIdx.x << 3) + (threadIdx.x >> 5);
  int lane = threadIdx.x & 31;
  int ms = wid / ntiles_n;
  int nt = wid - ms * ntiles_n;
  if (ms >= msuper) return;            // wave-uniform exit (EXEC stays all-ones)

  int HWl  = H * Wd;
  int Npix = Bn * HWl;
  int p = nt * 16 + (lane & 15);       // pixel column owned by this lane (B/D: N = lane%16)
  bool pvalid = p < Npix;
  int bb = 0, h = 0, w = 0;
  if (pvalid) { bb = p / HWl; int r = p - bb * HWl; h = r / Wd; w = r - h * Wd; }
  const float* xb = X + (size_t)bb * Cin * HWl;
  int khalf = (lane < 16) ? 0 : 16;    // B: lanes 0-15 hold K 0..15, lanes 16-31 hold K 16..31
  int icmax = Cin - 1;

  v8f acc[MB];
  #pragma unroll
  for (int i = 0; i < MB; ++i) acc[i] = (v8f){0.f, 0.f, 0.f, 0.f, 0.f, 0.f, 0.f, 0.f};

  const _Float16* apb = Ap + (size_t)(ms * MB) * nchunk * 512 + (size_t)lane * 16;

  int c = 0;  // running chunk index
  #pragma unroll
  for (int tap = 0; tap < 9; ++tap) {
    const int dh = tap / 3 - 1;
    const int dw = tap % 3 - 1;
    int hh = h + dh;
    int ww = w + dw;
    bool ok = pvalid && ((unsigned)hh < (unsigned)H) && ((unsigned)ww < (unsigned)Wd);
    const float* src = xb + (ok ? (hh * Wd + ww) : 0);
    float vm = ok ? 1.f : 0.f;         // branchless SAME zero padding

    for (int cc = 0; cc < cpt; ++cc, ++c) {
      int ic0 = cc * 32 + khalf;
      const float* src2 = src + (size_t)(unsigned)ic0 * (unsigned)HWl;
      v16h bf;
      #pragma unroll
      for (int j = 0; j < 16; ++j) {
        float v;
        if (CLAMP) {
          int ic = ic0 + j;
          ic = ic < icmax ? ic : icmax;      // channel pad: weight is zero there
          v = src[(size_t)(unsigned)ic * (unsigned)HWl];
        } else {
          v = src2[(size_t)((unsigned)j * (unsigned)HWl)];  // j*HW: uniform, hoisted
        }
        bf[j] = (_Float16)(v * vm);
      }
      #pragma unroll
      for (int mb = 0; mb < MB; ++mb) {
        v16h af = *(const v16h*)(apb + (size_t)(mb * nchunk + c) * 512);
        acc[mb] = __builtin_amdgcn_wmma_f32_16x16x32_f16(
            /*neg_a=*/false, af, /*neg_b=*/false, bf,
            /*c_mod=*/(short)0, acc[mb], /*reuse_a=*/false, /*reuse_b=*/false);
      }
    }
  }

  // ---- epilogue: bias + ReLU, D layout: VGPR i -> M = i + (lane<16?0:8) ----
  if (pvalid) {
    int rbase = (lane < 16) ? 0 : 8;
    int hw = p - bb * HWl;
    #pragma unroll
    for (int mb = 0; mb < MB; ++mb) {
      int oc0 = (ms * MB + mb) * 16 + rbase;
      #pragma unroll
      for (int i = 0; i < 8; ++i) {
        int oc = oc0 + i;
        if (oc < Cout) {
          float v = acc[mb][i] + bias[oc];
          v = fmaxf(v, 0.f);
          Y[((size_t)bb * Cout + oc) * HWl + hw] = v;
        }
      }
    }
  }
}

// ---------------------------------------------------------------------------
// host-side conv driver: pack weights then run WMMA conv (dispatch MB/CLAMP)
// ---------------------------------------------------------------------------
static void run_conv(hipStream_t s, const float* X, const float* W, const float* B,
                     float* Y, _Float16* Apack,
                     int Bn, int Cin, int Cout, int H, int Wd) {
  int cpt = (Cin + 31) / 32;           // chunks (of 32 channels) per tap
  int nchunk = 9 * cpt;
  int mb = (Cout + 15) / 16; if (mb > 4) mb = 4;
  if (mb == 3) mb = 2;                 // keep MB in {1,2,4}
  int msuper = (Cout + 16 * mb - 1) / (16 * mb);
  int mt_cnt = msuper * mb;
  long ptot = (long)mt_cnt * nchunk * 512;
  pack_w_kernel<<<grid_for(ptot), TPB, 0, s>>>(W, Apack, Cout, Cin, cpt, mt_cnt, nchunk);
  int Npix = Bn * H * Wd;
  int ntn  = (Npix + 15) / 16;
  long waves = (long)msuper * ntn;
  int blocks = (int)((waves + 7) / 8);
  bool clamp = (Cin & 31) != 0;
  #define LAUNCH(MBv, CLv) conv3x3_wmma_kernel<MBv, CLv><<<blocks, 256, 0, s>>>( \
      X, Apack, B, Y, Bn, Cin, Cout, H, Wd, cpt, nchunk, ntn, msuper)
  if (clamp) {
    if (mb == 1)      LAUNCH(1, true);
    else if (mb == 2) LAUNCH(2, true);
    else              LAUNCH(4, true);
  } else {
    if (mb == 1)      LAUNCH(1, false);
    else if (mb == 2) LAUNCH(2, false);
    else              LAUNCH(4, false);
  }
  #undef LAUNCH
}

// ---------------------------------------------------------------------------
extern "C" void kernel_launch(void* const* d_in, const int* in_sizes, int n_in,
                              void* d_out, int out_size, void* d_ws, size_t ws_size,
                              hipStream_t stream) {
  (void)in_sizes; (void)n_in; (void)out_size; (void)ws_size;
  const float* G_prev = (const float*)d_in[0];
  const float* G_cur  = (const float*)d_in[1];
  const float* I_prev = (const float*)d_in[2];
  // params flattened leaf-order: 11 stages x 3 convs x (w, b)
  const float* Wp[11][3]; const float* Bp[11][3];
  for (int st = 0; st < 11; ++st)
    for (int c = 0; c < 3; ++c) {
      Wp[st][c] = (const float*)d_in[3 + st * 6 + c * 2];
      Bp[st][c] = (const float*)d_in[3 + st * 6 + c * 2 + 1];
    }
  const int Bn = 2;
  const int HW256 = 256 * 256;

  float* ws = (float*)d_ws;
  size_t off = 0;
  auto alloc = [&](size_t n) { float* p = ws + off; off += (n + 63) & ~(size_t)63; return p; };
  float* bufA = alloc(4194304);        // [2,32,256,256] max
  float* bufB = alloc(4194304);
  float* s64  = alloc(2097152);        // d64  skip [2,64,128,128]
  float* s128 = alloc(1048576);        // d128 skip [2,128,64,64]
  float* s256 = alloc(524288);         // d256 skip [2,256,32,32]
  float* s512 = alloc(262144);         // d512 skip [2,512,16,16]
  float* u64b = alloc(2097152);        // u64       [2,64,128,128]
  float* khb  = alloc(2228224);        // [2,17,256,256]
  float* kwb  = alloc(2228224);
  _Float16* Apack = (_Float16*)alloc(1300000);  // 2.6M halves: max packed weights

  // ---- encoder ----
  concat_kernel<<<grid_for((long)Bn * 6 * HW256), TPB, 0, stream>>>(G_prev, G_cur, bufA, Bn, HW256);
  run_conv(stream, bufA, Wp[0][0], Bp[0][0], bufB, Apack, Bn,   6,  32, 256, 256);
  run_conv(stream, bufB, Wp[0][1], Bp[0][1], bufA, Apack, Bn,  32,  32, 256, 256);
  run_conv(stream, bufA, Wp[0][2], Bp[0][2], bufB, Apack, Bn,  32,  32, 256, 256);
  avgpool2_kernel<<<grid_for((long)Bn * 32 * 128 * 128), TPB, 0, stream>>>(bufB, bufA, Bn * 32, 256, 256);

  run_conv(stream, bufA, Wp[1][0], Bp[1][0], bufB, Apack, Bn,  32,  64, 128, 128);
  run_conv(stream, bufB, Wp[1][1], Bp[1][1], bufA, Apack, Bn,  64,  64, 128, 128);
  run_conv(stream, bufA, Wp[1][2], Bp[1][2], s64,  Apack, Bn,  64,  64, 128, 128);
  avgpool2_kernel<<<grid_for((long)Bn * 64 * 64 * 64), TPB, 0, stream>>>(s64, bufA, Bn * 64, 128, 128);

  run_conv(stream, bufA, Wp[2][0], Bp[2][0], bufB, Apack, Bn,  64, 128, 64, 64);
  run_conv(stream, bufB, Wp[2][1], Bp[2][1], bufA, Apack, Bn, 128, 128, 64, 64);
  run_conv(stream, bufA, Wp[2][2], Bp[2][2], s128, Apack, Bn, 128, 128, 64, 64);
  avgpool2_kernel<<<grid_for((long)Bn * 128 * 32 * 32), TPB, 0, stream>>>(s128, bufA, Bn * 128, 64, 64);

  run_conv(stream, bufA, Wp[3][0], Bp[3][0], bufB, Apack, Bn, 128, 256, 32, 32);
  run_conv(stream, bufB, Wp[3][1], Bp[3][1], bufA, Apack, Bn, 256, 256, 32, 32);
  run_conv(stream, bufA, Wp[3][2], Bp[3][2], s256, Apack, Bn, 256, 256, 32, 32);
  avgpool2_kernel<<<grid_for((long)Bn * 256 * 16 * 16), TPB, 0, stream>>>(s256, bufA, Bn * 256, 32, 32);

  run_conv(stream, bufA, Wp[4][0], Bp[4][0], bufB, Apack, Bn, 256, 512, 16, 16);
  run_conv(stream, bufB, Wp[4][1], Bp[4][1], bufA, Apack, Bn, 512, 512, 16, 16);
  run_conv(stream, bufA, Wp[4][2], Bp[4][2], s512, Apack, Bn, 512, 512, 16, 16);
  avgpool2_kernel<<<grid_for((long)Bn * 512 * 8 * 8), TPB, 0, stream>>>(s512, bufA, Bn * 512, 16, 16);

  run_conv(stream, bufA, Wp[5][0], Bp[5][0], bufB, Apack, Bn, 512, 512, 8, 8);
  run_conv(stream, bufB, Wp[5][1], Bp[5][1], bufA, Apack, Bn, 512, 512, 8, 8);
  run_conv(stream, bufA, Wp[5][2], Bp[5][2], bufB, Apack, Bn, 512, 512, 8, 8);

  // ---- decoder ----
  upsample2_kernel<<<grid_for((long)Bn * 512 * 16 * 16), TPB, 0, stream>>>(bufB, s512, bufA, Bn * 512, 16, 16, 1);
  run_conv(stream, bufA, Wp[6][0], Bp[6][0], bufB, Apack, Bn, 512, 256, 16, 16);
  run_conv(stream, bufB, Wp[6][1], Bp[6][1], bufA, Apack, Bn, 256, 256, 16, 16);
  run_conv(stream, bufA, Wp[6][2], Bp[6][2], bufB, Apack, Bn, 256, 256, 16, 16);
  upsample2_kernel<<<grid_for((long)Bn * 256 * 32 * 32), TPB, 0, stream>>>(bufB, s256, bufA, Bn * 256, 32, 32, 1);

  run_conv(stream, bufA, Wp[7][0], Bp[7][0], bufB, Apack, Bn, 256, 128, 32, 32);
  run_conv(stream, bufB, Wp[7][1], Bp[7][1], bufA, Apack, Bn, 128, 128, 32, 32);
  run_conv(stream, bufA, Wp[7][2], Bp[7][2], bufB, Apack, Bn, 128, 128, 32, 32);
  upsample2_kernel<<<grid_for((long)Bn * 128 * 64 * 64), TPB, 0, stream>>>(bufB, s128, bufA, Bn * 128, 64, 64, 1);

  run_conv(stream, bufA, Wp[8][0], Bp[8][0], bufB, Apack, Bn, 128, 64, 64, 64);
  run_conv(stream, bufB, Wp[8][1], Bp[8][1], bufA, Apack, Bn,  64, 64, 64, 64);
  run_conv(stream, bufA, Wp[8][2], Bp[8][2], bufB, Apack, Bn,  64, 64, 64, 64);
  upsample2_kernel<<<grid_for((long)Bn * 64 * 128 * 128), TPB, 0, stream>>>(bufB, s64, u64b, Bn * 64, 128, 128, 1);

  // ---- kernel-prediction heads ----
  run_conv(stream, u64b, Wp[9][0], Bp[9][0], bufA, Apack, Bn, 64, 17, 128, 128);
  run_conv(stream, bufA, Wp[9][1], Bp[9][1], bufB, Apack, Bn, 17, 17, 128, 128);
  run_conv(stream, bufB, Wp[9][2], Bp[9][2], bufA, Apack, Bn, 17, 17, 128, 128);
  upsample2_kernel<<<grid_for((long)Bn * 17 * HW256), TPB, 0, stream>>>(bufA, bufA, khb, Bn * 17, 256, 256, 0);
  softmax_c_kernel<<<grid_for((long)Bn * HW256), TPB, 0, stream>>>(khb, Bn, 17, HW256);

  run_conv(stream, u64b, Wp[10][0], Bp[10][0], bufA, Apack, Bn, 64, 17, 128, 128);
  run_conv(stream, bufA, Wp[10][1], Bp[10][1], bufB, Apack, Bn, 17, 17, 128, 128);
  run_conv(stream, bufB, Wp[10][2], Bp[10][2], bufA, Apack, Bn, 17, 17, 128, 128);
  upsample2_kernel<<<grid_for((long)Bn * 17 * HW256), TPB, 0, stream>>>(bufA, bufA, kwb, Bn * 17, 256, 256, 0);
  softmax_c_kernel<<<grid_for((long)Bn * HW256), TPB, 0, stream>>>(kwb, Bn, 17, HW256);

  // ---- separable adaptive filtering of I_prev ----
  sep_conv_kernel<<<grid_for((long)Bn * 3 * HW256), TPB, 0, stream>>>(
      I_prev, khb, kwb, (float*)d_out, Bn, 3, 256, 256);
}
